// Attention_58634893525740
// MI455X (gfx1250) — matrix operand order repaired and checked by reading, not compiled, not run on hardware
//
#include <hip/hip_runtime.h>
#include <hip/hip_bf16.h>

// MI455X / gfx1250 — wave32, WMMA bf16 16x16x32 (f32 acc), async global->LDS double buffering.
#define DIMC 768
#define NH   12
#define HD   64

typedef __attribute__((ext_vector_type(16))) __bf16        v16bf;
typedef __attribute__((ext_vector_type(8)))  float         v8f;
typedef __attribute__((ext_vector_type(8)))  unsigned int  v8u;

union FragBF { v8u u; v16bf bf; };

__device__ __forceinline__ unsigned int packbf2(float a, float b) {
  unsigned int ua = __builtin_bit_cast(unsigned int, a);
  unsigned int ub = __builtin_bit_cast(unsigned int, b);
  ua = (ua + 0x7FFFu + ((ua >> 16) & 1u)) >> 16;   // round-to-nearest-even
  ub = (ub + 0x7FFFu + ((ub >> 16) & 1u)) >> 16;
  return (ub << 16) | (ua & 0xFFFFu);
}
__device__ __forceinline__ unsigned short f2bf(float a) {
  unsigned int ua = __builtin_bit_cast(unsigned int, a);
  return (unsigned short)((ua + 0x7FFFu + ((ua >> 16) & 1u)) >> 16);
}
// A-matrix 16-bit 16x32 K-offsets: lanes 0-15 hold K=0..7,16..23; lanes 16-31 hold K=8..15,24..31
__device__ __forceinline__ int koffA(int v, int hi8) {
  return (v < 4) ? (hi8 + 2 * v) : (16 + hi8 + 2 * (v - 4));
}
__device__ __forceinline__ v8f wmma_bf16(const FragBF& a, const FragBF& b, v8f c) {
  return __builtin_amdgcn_wmma_f32_16x16x32_bf16(false, a.bf, false, b.bf,
                                                 (short)0, c, false, false);
}

// --- CDNA5 async global->LDS (ASYNCcnt-tracked, GVS addressing: saddr + 32-bit vaddr) ---
__device__ __forceinline__ void async_load_b128(unsigned lds_byte, unsigned goff_byte,
                                                const void* sbase) {
  asm volatile("global_load_async_to_lds_b128 %0, %1, %2"
               :: "v"(lds_byte), "v"(goff_byte), "s"(sbase) : "memory");
}
__device__ __forceinline__ void wait_async_le2() {
  asm volatile("s_wait_asynccnt 0x2" ::: "memory");
}
__device__ __forceinline__ void wait_async_le0() {
  asm volatile("s_wait_asynccnt 0x0" ::: "memory");
}

// ---------------- Kernel 1: QKV GEMM (x[8192,768] x qkv_w^T[768,2304]) ----------------
// Routes outputs into bf16 Q[bh][n][d] (pre-scaled), K[bh][n][d], Vt[bh][d][n].
__global__ __launch_bounds__(256) void qkv_gemm(const float* __restrict__ x,
                                                const float* __restrict__ qkv_w,
                                                unsigned short* __restrict__ Qb,
                                                unsigned short* __restrict__ Kb,
                                                unsigned short* __restrict__ Vt) {
  __shared__ __align__(16) unsigned short As[128 * 32];
  __shared__ __align__(16) unsigned short Ws[64 * 32];
  const int t = threadIdx.x, lane = t & 31, w = t >> 5;
  const int ln = lane & 15, hi8 = (lane & 16) ? 8 : 0;
  const int wm = w & 3, wn = w >> 2;
  const int m0 = blockIdx.x * 128, o0 = blockIdx.y * 64;

  v8f c[2][2] = {};
  for (int k0 = 0; k0 < DIMC; k0 += 32) {
    {  // A tile 128x32 f32 -> bf16
      int row = t >> 1, c0 = (t & 1) * 16;
      const float4* src = (const float4*)(x + (size_t)(m0 + row) * DIMC + k0 + c0);
      unsigned int* dst = (unsigned int*)&As[row * 32 + c0];
#pragma unroll
      for (int i = 0; i < 4; ++i) {
        float4 f = src[i];
        dst[2 * i] = packbf2(f.x, f.y);
        dst[2 * i + 1] = packbf2(f.z, f.w);
      }
    }
    {  // W tile 64x32 f32 -> bf16
      int row = t >> 2, c0 = (t & 3) * 8;
      const float4* src = (const float4*)(qkv_w + (size_t)(o0 + row) * DIMC + k0 + c0);
      unsigned int* dst = (unsigned int*)&Ws[row * 32 + c0];
#pragma unroll
      for (int i = 0; i < 2; ++i) {
        float4 f = src[i];
        dst[2 * i] = packbf2(f.x, f.y);
        dst[2 * i + 1] = packbf2(f.z, f.w);
      }
    }
    __syncthreads();
    FragBF a[2], bfr[2];
#pragma unroll
    for (int mi = 0; mi < 2; ++mi) {
      int row = wm * 32 + mi * 16 + ln;
#pragma unroll
      for (int v = 0; v < 8; ++v)
        a[mi].u[v] = *(const unsigned int*)&As[row * 32 + koffA(v, hi8)];
    }
#pragma unroll
    for (int oi = 0; oi < 2; ++oi) {
      int col = wn * 32 + oi * 16 + ln;
      int kb = (lane & 16) ? 16 : 0;
#pragma unroll
      for (int v = 0; v < 8; ++v)
        bfr[oi].u[v] = *(const unsigned int*)&Ws[col * 32 + koffA(v, hi8)];
    }
#pragma unroll
    for (int mi = 0; mi < 2; ++mi)
#pragma unroll
      for (int oi = 0; oi < 2; ++oi) c[mi][oi] = wmma_bf16(a[mi], bfr[oi], c[mi][oi]);
    __syncthreads();
  }
  // epilogue: route columns to Q (scaled), K, Vt
#pragma unroll
  for (int mi = 0; mi < 2; ++mi) {
#pragma unroll
    for (int oi = 0; oi < 2; ++oi) {
      int o = o0 + wn * 32 + oi * 16 + ln;
      int three = o / 768, rem = o % 768;
      int h = rem / 64, d = rem % 64;
#pragma unroll
      for (int r = 0; r < 8; ++r) {
        int m = m0 + wm * 32 + mi * 16 + r + hi8;
        int bi = m >> 10, n = m & 1023;
        float val = c[mi][oi][r];
        if (three == 0) {
          Qb[((size_t)((bi * NH + h) << 10) + n) * HD + d] = f2bf(val * 0.125f);
        } else if (three == 1) {
          Kb[((size_t)((bi * NH + h) << 10) + n) * HD + d] = f2bf(val);
        } else {
          Vt[(((size_t)(bi * NH + h) * HD + d) << 10) + n] = f2bf(val);
        }
      }
    }
  }
}

// ---------------- Kernel 2: flash attention, one (b,h) x 128 queries per WG ----------------
// K/V^T blocks streamed via async global->LDS with double buffering (ASYNCcnt pipelined).
__global__ __launch_bounds__(256) void attn(const unsigned short* __restrict__ Qb,
                                            const unsigned short* __restrict__ Kb,
                                            const unsigned short* __restrict__ Vt,
                                            const float* __restrict__ bias,
                                            unsigned short* __restrict__ Ob) {
  __shared__ __align__(16) unsigned short Kt[2][32 * 64];   // keys block [j][d]
  __shared__ __align__(16) unsigned short Vs[2][64 * 32];   // V^T block  [d][j]
  __shared__ __align__(16) unsigned short Pw[8][16 * 32];   // per-wave P tile
  const int t = threadIdx.x, lane = t & 31, w = t >> 5;
  const int ln = lane & 15, hi8 = (lane & 16) ? 8 : 0;
  const int bh = blockIdx.y;
  const int h = bh % NH;
  const int q0 = blockIdx.x * 128 + w * 16;

  const unsigned short* kbase = Kb + ((size_t)(bh << 10)) * HD;
  const unsigned short* vbase = Vt + (((size_t)bh * HD) << 10);
  const int vrow = t >> 2, vc0 = (t & 3) * 8;
  const unsigned ldsK[2] = {(unsigned)(uintptr_t)&Kt[0][t * 8],
                            (unsigned)(uintptr_t)&Kt[1][t * 8]};
  const unsigned ldsV[2] = {(unsigned)(uintptr_t)&Vs[0][vrow * 32 + vc0],
                            (unsigned)(uintptr_t)&Vs[1][vrow * 32 + vc0]};

  auto issueKV = [&](int kbn, int buf) {
    unsigned gk = (unsigned)((kbn * HD + t * 8) * 2);
    unsigned gv = (unsigned)(((vrow << 10) + kbn + vc0) * 2);
    async_load_b128(ldsK[buf], gk, kbase);
    async_load_b128(ldsV[buf], gv, vbase);
  };

  FragBF qa[2];  // Q rows, K-dim 64 -> two K=32 fragments
  {
    const unsigned short* qrow = Qb + ((size_t)(bh << 10) + q0 + ln) * HD;
#pragma unroll
    for (int f = 0; f < 2; ++f)
#pragma unroll
      for (int v = 0; v < 8; ++v)
        qa[f].u[v] = *(const unsigned int*)(qrow + f * 32 + koffA(v, hi8));
  }

  v8f oacc[4] = {};
  float mrow[8], lrow[8];
#pragma unroll
  for (int r = 0; r < 8; ++r) { mrow[r] = -1e30f; lrow[r] = 0.f; }
  const float* biasH = bias + (size_t)h * 1024 * 1024;

  issueKV(0, 0);  // prologue: block 0 -> buffer 0
  for (int kb = 0; kb < 1024; kb += 32) {
    const int buf = (kb >> 5) & 1;
    __syncthreads();  // all waves finished reading buf^1 (previous block)
    if (kb + 32 < 1024) {
      issueKV(kb + 32, buf ^ 1);  // prefetch next block into other buffer
      wait_async_le2();           // in-order per wave: current block complete
    } else {
      wait_async_le0();
    }
    __syncthreads();  // current buffer visible to all waves

    v8f s[2];
#pragma unroll
    for (int jt = 0; jt < 2; ++jt) {  // two 16-key tiles
      FragBF k0f, k1f;
      int col = jt * 16 + ln;
      int dlo = (lane & 16) ? 16 : 0;
#pragma unroll
      for (int v = 0; v < 8; ++v) {
        k0f.u[v] = *(const unsigned int*)&Kt[buf][col * 64 + koffA(v, hi8)];
        k1f.u[v] = *(const unsigned int*)&Kt[buf][col * 64 + 32 + koffA(v, hi8)];
      }
      v8f z = {};
      z = wmma_bf16(qa[0], k0f, z);
      z = wmma_bf16(qa[1], k1f, z);
#pragma unroll
      for (int r = 0; r < 8; ++r)  // bias in C-layout
        z[r] = z[r] + biasH[(size_t)(q0 + r + hi8) * 1024 + kb + jt * 16 + ln];
      s[jt] = z;
    }

    // online softmax (rows live across 16-lane halves)
#pragma unroll
    for (int r = 0; r < 8; ++r) {
      float mx = fmaxf(s[0][r], s[1][r]);
#pragma unroll
      for (int off = 8; off >= 1; off >>= 1) mx = fmaxf(mx, __shfl_xor(mx, off, 32));
      float mnew = fmaxf(mrow[r], mx);
      float alpha = __expf(mrow[r] - mnew);
      float p0 = __expf(s[0][r] - mnew);
      float p1 = __expf(s[1][r] - mnew);
      s[0][r] = p0;
      s[1][r] = p1;
      float rs = p0 + p1;
#pragma unroll
      for (int off = 8; off >= 1; off >>= 1) rs += __shfl_xor(rs, off, 32);
      lrow[r] = lrow[r] * alpha + rs;
      mrow[r] = mnew;
#pragma unroll
      for (int dt = 0; dt < 4; ++dt) oacc[dt][r] = oacc[dt][r] * alpha;
    }

    // P (C-layout) -> per-wave LDS tile -> A-layout fragment
#pragma unroll
    for (int jt = 0; jt < 2; ++jt)
#pragma unroll
      for (int r = 0; r < 8; ++r)
        Pw[w][(r + hi8) * 32 + jt * 16 + ln] = f2bf(s[jt][r]);
    __builtin_amdgcn_wave_barrier();  // HW keeps same-wave DS ops in order

    FragBF pa;
#pragma unroll
    for (int v = 0; v < 8; ++v)
      pa.u[v] = *(const unsigned int*)&Pw[w][ln * 32 + koffA(v, hi8)];
#pragma unroll
    for (int dt = 0; dt < 4; ++dt) {
      FragBF vb;
      int d = dt * 16 + ln;
      int jlo = (lane & 16) ? 16 : 0;
#pragma unroll
      for (int v = 0; v < 8; ++v)
        vb.u[v] = *(const unsigned int*)&Vs[buf][d * 32 + koffA(v, hi8)];
      oacc[dt] = wmma_bf16(pa, vb, oacc[dt]);
    }
  }

  // normalize, store bf16 O[b][n][h][d] (head-concat rows for the final GEMM)
  const int bi = bh / NH;
#pragma unroll
  for (int r = 0; r < 8; ++r) {
    float inv = 1.0f / lrow[r];
    int n = q0 + r + hi8;
#pragma unroll
    for (int dt = 0; dt < 4; ++dt) {
      int d = dt * 16 + ln;
      Ob[(((size_t)(bi << 10) + n) * NH + h) * HD + d] = f2bf(oacc[dt][r] * inv);
    }
  }
}

// ---------------- Kernel 3: output projection (+bias), f32 out ----------------
// A-tile (bf16 attention output) streamed via async global->LDS double buffering.
__global__ __launch_bounds__(256) void proj_gemm(const unsigned short* __restrict__ Ob,
                                                 const float* __restrict__ proj_w,
                                                 const float* __restrict__ proj_b,
                                                 float* __restrict__ out) {
  __shared__ __align__(16) unsigned short As[2][128 * 32];
  __shared__ __align__(16) unsigned short Ws[64 * 32];
  const int t = threadIdx.x, lane = t & 31, w = t >> 5;
  const int ln = lane & 15, hi8 = (lane & 16) ? 8 : 0;
  const int wm = w & 3, wn = w >> 2;
  const int m0 = blockIdx.x * 128, o0 = blockIdx.y * 64;

  const unsigned short* abase = Ob + (size_t)m0 * DIMC;
  const int arow = t >> 1, ac0 = (t & 1) * 16;
  const unsigned ldsA[2] = {(unsigned)(uintptr_t)&As[0][arow * 32 + ac0],
                            (unsigned)(uintptr_t)&As[1][arow * 32 + ac0]};
  auto issueA = [&](int k0n, int buf) {
    unsigned ga = (unsigned)((arow * DIMC + k0n + ac0) * 2);
    async_load_b128(ldsA[buf], ga, abase);
    async_load_b128(ldsA[buf] + 16u, ga + 16u, abase);
  };

  v8f c[2][2] = {};
  issueA(0, 0);
  for (int k0 = 0; k0 < DIMC; k0 += 32) {
    const int buf = (k0 >> 5) & 1;
    __syncthreads();  // Ws and As[buf^1] free
    // W tile f32 -> bf16 (register staged; overlaps with async A stream)
    int row = t >> 2, c0 = (t & 3) * 8;
    const float4* src = (const float4*)(proj_w + (size_t)(o0 + row) * DIMC + k0 + c0);
    float4 f0 = src[0], f1 = src[1];
    if (k0 + 32 < DIMC) {
      issueA(k0 + 32, buf ^ 1);
      wait_async_le2();
    } else {
      wait_async_le0();
    }
    unsigned int* dst = (unsigned int*)&Ws[row * 32 + c0];
    dst[0] = packbf2(f0.x, f0.y);
    dst[1] = packbf2(f0.z, f0.w);
    dst[2] = packbf2(f1.x, f1.y);
    dst[3] = packbf2(f1.z, f1.w);
    __syncthreads();

    FragBF a[2], bfr[2];
#pragma unroll
    for (int mi = 0; mi < 2; ++mi) {
      int arw = wm * 32 + mi * 16 + ln;
#pragma unroll
      for (int v = 0; v < 8; ++v)
        a[mi].u[v] = *(const unsigned int*)&As[buf][arw * 32 + koffA(v, hi8)];
    }
#pragma unroll
    for (int oi = 0; oi < 2; ++oi) {
      int col = wn * 32 + oi * 16 + ln;
      int kb = (lane & 16) ? 16 : 0;
#pragma unroll
      for (int v = 0; v < 8; ++v)
        bfr[oi].u[v] = *(const unsigned int*)&Ws[col * 32 + koffA(v, hi8)];
    }
#pragma unroll
    for (int mi = 0; mi < 2; ++mi)
#pragma unroll
      for (int oi = 0; oi < 2; ++oi) c[mi][oi] = wmma_bf16(a[mi], bfr[oi], c[mi][oi]);
  }
#pragma unroll
  for (int mi = 0; mi < 2; ++mi) {
#pragma unroll
    for (int oi = 0; oi < 2; ++oi) {
      int o = o0 + wn * 32 + oi * 16 + ln;
      float pb = proj_b[o];
#pragma unroll
      for (int r = 0; r < 8; ++r) {
        int m = m0 + wm * 32 + mi * 16 + r + hi8;
        out[(size_t)m * DIMC + o] = c[mi][oi][r] + pb;
      }
    }
  }
}

extern "C" void kernel_launch(void* const* d_in, const int* in_sizes, int n_in,
                              void* d_out, int out_size, void* d_ws, size_t ws_size,
                              hipStream_t stream) {
  (void)in_sizes; (void)n_in; (void)out_size; (void)ws_size;
  const float* x      = (const float*)d_in[0];
  const float* bias   = (const float*)d_in[1];
  const float* qkv_w  = (const float*)d_in[2];
  const float* proj_w = (const float*)d_in[3];
  const float* proj_b = (const float*)d_in[4];
  float* out = (float*)d_out;

  char* ws = (char*)d_ws;  // 48 MB of bf16 intermediates
  unsigned short* Qb = (unsigned short*)(ws);
  unsigned short* Kb = (unsigned short*)(ws + 12582912);
  unsigned short* Vt = (unsigned short*)(ws + 25165824);
  unsigned short* Ob = (unsigned short*)(ws + 37748736);

  qkv_gemm<<<dim3(64, 36), 256, 0, stream>>>(x, qkv_w, Qb, Kb, Vt);
  attn<<<dim3(8, 96), 256, 0, stream>>>(Qb, Kb, Vt, bias, Ob);
  proj_gemm<<<dim3(64, 12), 256, 0, stream>>>(Ob, proj_w, proj_b, out);
}
